// CrossAttention_1881195675984
// MI455X (gfx1250) — matrix operand order repaired and checked by reading, not compiled, not run on hardware
//
#include <hip/hip_runtime.h>

typedef __bf16 bf16_t;
typedef bf16_t v16bf __attribute__((ext_vector_type(16)));
typedef float  v8f   __attribute__((ext_vector_type(8)));
typedef unsigned short u16;

#define DIM      1024
#define HEADS    16
#define HEAD_DIM 64
#define BATCH    4
#define NX       1024
#define NY       1024
#define SCALE    0.125f   // HEAD_DIM^-0.5

union FragBF {
  uint4 q[2];
  u16   s[16];
  v16bf v;
};

__device__ __forceinline__ u16 f2bf(float x) {
  unsigned int u = __float_as_uint(x);
  u += 0x7FFFu + ((u >> 16) & 1u);          // round-to-nearest-even
  return (u16)(u >> 16);
}

// ---- WMMA fragment loaders (wave32, v_wmma_f32_16x16x32_bf16) -------------
// A (16x32 bf16) from a row-major [16 x ld] bf16 tile (ISA 7.12.2 layout).
__device__ __forceinline__ v16bf load_frag_a(const u16* base,
                                             int ld, int k0, int lane) {
  const int m = lane & 15, half = lane >> 4;
  const u16* p = base + m * ld + k0 + half * 8;
  FragBF f;
  f.q[0] = *(const uint4*)(p);
  f.q[1] = *(const uint4*)(p + 16);
  return f.v;
}

// B (32x16 bf16) where B[k][n] = W[n][k], W row-major [16 x ld].
__device__ __forceinline__ v16bf load_frag_b(const u16* base,
                                             int ld, int k0, int lane) {
  const int n = lane & 15, half = lane >> 4;
  const u16* p = base + n * ld + k0 + half * 8;
  FragBF f;
  f.q[0] = *(const uint4*)(p);
  f.q[1] = *(const uint4*)(p + 16);
  return f.v;
}

__device__ __forceinline__ v8f wmma_bf16(v16bf a, v16bf b, v8f c) {
  return __builtin_amdgcn_wmma_f32_16x16x32_bf16(false, a, false, b,
                                                 (short)0, c, false, false);
}

// 32x64 macro-tile per wave: C[2][4] += A(2 tiles) * B(4 tiles)^T over K.
// A rows at A0/A1 (row-major, ld=K); B is 4 consecutive 16-row tiles at B0.
__device__ __forceinline__ void gemm_block_2x4(const u16* __restrict__ A0,
                                               const u16* __restrict__ A1,
                                               const u16* __restrict__ B0,
                                               int K, int lane, v8f acc[2][4]) {
  for (int k0 = 0; k0 < K; k0 += 32) {
    v16bf a0 = load_frag_a(A0, K, k0, lane);
    v16bf a1 = load_frag_a(A1, K, k0, lane);
    v16bf b0 = load_frag_b(B0 + 0 * 16 * K, K, k0, lane);
    v16bf b1 = load_frag_b(B0 + 1 * 16 * K, K, k0, lane);
    v16bf b2 = load_frag_b(B0 + 2 * 16 * K, K, k0, lane);
    v16bf b3 = load_frag_b(B0 + 3 * 16 * K, K, k0, lane);
    acc[0][0] = wmma_bf16(a0, b0, acc[0][0]);
    acc[0][1] = wmma_bf16(a0, b1, acc[0][1]);
    acc[0][2] = wmma_bf16(a0, b2, acc[0][2]);
    acc[0][3] = wmma_bf16(a0, b3, acc[0][3]);
    acc[1][0] = wmma_bf16(a1, b0, acc[1][0]);
    acc[1][1] = wmma_bf16(a1, b1, acc[1][1]);
    acc[1][2] = wmma_bf16(a1, b2, acc[1][2]);
    acc[1][3] = wmma_bf16(a1, b3, acc[1][3]);
  }
}

// ---------------------------------------------------------------------------
__global__ void cvt_f32_bf16(const float* __restrict__ in,
                             u16* __restrict__ out, int n) {
  int i = blockIdx.x * blockDim.x + threadIdx.x;
  if (i < n) out[i] = f2bf(__builtin_nontemporal_load(in + i));
}

// q = x @ Wq^T  -> q_bf [B,H,Nx,64] bf16.  Wave: 32x64 macro-tile.
__global__ __launch_bounds__(256) void qproj_kernel(
    const u16* __restrict__ x_bf, const u16* __restrict__ wq_bf,
    u16* __restrict__ q_bf) {
  const int lane = threadIdx.x & 31;
  const int task = blockIdx.x * 8 + (threadIdx.x >> 5);   // 128 x 16
  const int rb = task >> 4, cb = task & 15;
  v8f acc[2][4] = {};
  gemm_block_2x4(x_bf + (size_t)(rb * 32) * DIM,
                 x_bf + (size_t)(rb * 32 + 16) * DIM,
                 wq_bf + (size_t)(cb * 64) * DIM, DIM, lane, acc);
  const int n = lane & 15, half = lane >> 4;
#pragma unroll
  for (int tt = 0; tt < 2; ++tt)
#pragma unroll
    for (int cc = 0; cc < 4; ++cc) {
      const int j = cb * 64 + cc * 16 + n, h = j >> 6, d = j & 63;
#pragma unroll
      for (int r = 0; r < 8; ++r) {
        const int i = rb * 32 + tt * 16 + r + 8 * half;
        const int b = i >> 10, ix = i & 1023;
        q_bf[(((b * HEADS + h) * NX) + ix) * HEAD_DIM + d] = f2bf(acc[tt][cc][r]);
      }
    }
}

// kv = y @ Wkv^T -> k_bf [B,H,Ny,64] bf16 ; vT_bf [B,H,64,Ny] bf16
__global__ __launch_bounds__(256) void kvproj_kernel(
    const u16* __restrict__ y_bf, const u16* __restrict__ wkv_bf,
    u16* __restrict__ k_bf, u16* __restrict__ vT_bf) {
  const int lane = threadIdx.x & 31;
  const int task = blockIdx.x * 8 + (threadIdx.x >> 5);   // 128 x 32
  const int rb = task >> 5, cb = task & 31;
  v8f acc[2][4] = {};
  gemm_block_2x4(y_bf + (size_t)(rb * 32) * DIM,
                 y_bf + (size_t)(rb * 32 + 16) * DIM,
                 wkv_bf + (size_t)(cb * 64) * DIM, DIM, lane, acc);
  const int n = lane & 15, half = lane >> 4;
#pragma unroll
  for (int tt = 0; tt < 2; ++tt)
#pragma unroll
    for (int cc = 0; cc < 4; ++cc) {
      const int j = cb * 64 + cc * 16 + n;
#pragma unroll
      for (int r = 0; r < 8; ++r) {
        const int i = rb * 32 + tt * 16 + r + 8 * half;
        const int b = i >> 10, iy = i & 1023;
        const u16 val = f2bf(acc[tt][cc][r]);
        if (j < DIM) {                          // K half (uniform per tile)
          const int h = j >> 6, d = j & 63;
          k_bf[(((b * HEADS + h) * NY) + iy) * HEAD_DIM + d] = val;
        } else {                                // V half, stored transposed
          const int j2 = j - DIM, h = j2 >> 6, d = j2 & 63;
          vT_bf[(((size_t)(b * HEADS + h) * HEAD_DIM) + d) * NY + iy] = val;
        }
      }
    }
}

// Fused attention: S = q k^T * scale ; softmax ; * weights -> attn (f32 out,
// non-temporal stream) ; O = attn @ v (bf16).  Block = (b, h, 16-row q tile).
__global__ __launch_bounds__(256) void attn_kernel(
    const u16* __restrict__ q_bf, const u16* __restrict__ k_bf,
    const u16* __restrict__ vT_bf, const float* __restrict__ weights,
    float* __restrict__ attn_out, u16* __restrict__ o_bf) {
  __shared__ float S[16][NY];      // 64 KB; later: bytes [0,32K) = bf16 P,
                                   //          bytes [32K,36K) = split-K scratch
  const int blk = blockIdx.x;
  const int qt = blk & 63;
  const int bh = blk >> 6;                          // b*H + h
  const int b = bh >> 4, h = bh & 15;
  const int w = threadIdx.x >> 5, lane = threadIdx.x & 31;
  const int n = lane & 15, half = lane >> 4;

  const u16* qbase = q_bf + ((size_t)bh * NX + qt * 16) * HEAD_DIM;
  const u16* kbase = k_bf + (size_t)bh * NY * HEAD_DIM;

  // Phase 1: S row-block; hoist the K-invariant Q fragments.
  {
    const v16bf aq0 = load_frag_a(qbase, HEAD_DIM, 0, lane);
    const v16bf aq1 = load_frag_a(qbase, HEAD_DIM, 32, lane);
    for (int jt = w * 8; jt < w * 8 + 8; ++jt) {
      const u16* kt = kbase + (size_t)jt * 16 * HEAD_DIM;
      v16bf b0 = load_frag_b(kt, HEAD_DIM, 0, lane);
      v16bf b1 = load_frag_b(kt, HEAD_DIM, 32, lane);
      v8f acc = {};
      acc = wmma_bf16(aq0, b0, acc);
      acc = wmma_bf16(aq1, b1, acc);
#pragma unroll
      for (int r = 0; r < 8; ++r)
        S[r + 8 * half][jt * 16 + n] = acc[r] * SCALE;
    }
  }
  __syncthreads();

  // Phase 2: softmax (wave w owns rows 2w,2w+1); stage weighted probs in regs.
  const float* wbase = weights + ((size_t)b * NX + qt * 16) * NY;
  float* abase = attn_out + ((size_t)bh * NX + qt * 16) * NY;
  float pv[2][32];
#pragma unroll
  for (int rr = 0; rr < 2; ++rr) {
    const int row = w * 2 + rr;
#pragma unroll
    for (int j = 0; j < 32; ++j) pv[rr][j] = S[row][lane + 32 * j];
    float mx = -3.0e38f;
#pragma unroll
    for (int j = 0; j < 32; ++j) mx = fmaxf(mx, pv[rr][j]);
#pragma unroll
    for (int m = 16; m >= 1; m >>= 1) mx = fmaxf(mx, __shfl_xor(mx, m, 32));
    float sum = 0.f;
#pragma unroll
    for (int j = 0; j < 32; ++j) { pv[rr][j] = __expf(pv[rr][j] - mx); sum += pv[rr][j]; }
#pragma unroll
    for (int m = 16; m >= 1; m >>= 1) sum += __shfl_xor(sum, m, 32);
    const float inv = 1.0f / sum;
    const float* wrow = wbase + (size_t)row * NY;
#pragma unroll
    for (int j = 0; j < 32; ++j) pv[rr][j] *= inv * wrow[lane + 32 * j];
  }
  __syncthreads();                 // all f32 reads of S complete

  // Write attn (non-temporal, never re-read, > L2) and bf16 P into LDS.
  u16* P = (u16*)&S[0][0];
#pragma unroll
  for (int rr = 0; rr < 2; ++rr) {
    const int row = w * 2 + rr;
    float* arow = abase + (size_t)row * NY;
#pragma unroll
    for (int j = 0; j < 32; ++j) {
      const int c = lane + 32 * j;
      __builtin_nontemporal_store(pv[rr][j], arow + c);
      P[row * NY + c] = f2bf(pv[rr][j]);
    }
  }
  __syncthreads();

  // Phase 3: O = P @ V.  wave -> (col-tile ct, K-half kh); split-K pairs.
  const int ct = w & 3, kh = w >> 2;
  const u16* vbase = vT_bf + ((size_t)bh * HEAD_DIM + ct * 16) * NY;
  const u16* Pc = (const u16*)&S[0][0];
  v8f acc = {};
  for (int ks = 0; ks < 16; ++ks) {
    const int k0 = kh * 512 + ks * 32;
    v16bf af = load_frag_a(Pc, NY, k0, lane);        // ds_load_b128 x2
    v16bf bf = load_frag_b(vbase, NY, k0, lane);
    acc = wmma_bf16(af, bf, acc);
  }
  __syncthreads();                 // everyone done reading P
  float* red = (float*)((char*)&S[0][0] + 32 * 1024);   // 4 KB scratch
  if (kh == 1) {
#pragma unroll
    for (int r = 0; r < 8; ++r) red[(ct * 8 + r) * 32 + lane] = acc[r];
  }
  __syncthreads();
  if (kh == 0) {
#pragma unroll
    for (int r = 0; r < 8; ++r) acc[r] += red[(ct * 8 + r) * 32 + lane];
#pragma unroll
    for (int r = 0; r < 8; ++r) {
      const int m = r + 8 * half;
      const size_t i = (size_t)b * NX + qt * 16 + m;      // global row
      o_bf[i * DIM + h * HEAD_DIM + ct * 16 + n] = f2bf(acc[r]);
    }
  }
}

// out = o @ Wp^T + bp  (f32 result).  Wave: 32x64 macro-tile.
__global__ __launch_bounds__(256) void outproj_kernel(
    const u16* __restrict__ o_bf, const u16* __restrict__ wp_bf,
    const float* __restrict__ bp, float* __restrict__ out) {
  const int lane = threadIdx.x & 31;
  const int task = blockIdx.x * 8 + (threadIdx.x >> 5);   // 128 x 16
  const int rb = task >> 4, cb = task & 15;
  v8f acc[2][4] = {};
  gemm_block_2x4(o_bf + (size_t)(rb * 32) * DIM,
                 o_bf + (size_t)(rb * 32 + 16) * DIM,
                 wp_bf + (size_t)(cb * 64) * DIM, DIM, lane, acc);
  const int n = lane & 15, half = lane >> 4;
#pragma unroll
  for (int tt = 0; tt < 2; ++tt)
#pragma unroll
    for (int cc = 0; cc < 4; ++cc) {
      const int j = cb * 64 + cc * 16 + n;
      const float bias = bp[j];
#pragma unroll
      for (int r = 0; r < 8; ++r) {
        const int i = rb * 32 + tt * 16 + r + 8 * half;
        out[(size_t)i * DIM + j] = acc[tt][cc][r] + bias;
      }
    }
}

// ---------------------------------------------------------------------------
extern "C" void kernel_launch(void* const* d_in, const int* in_sizes, int n_in,
                              void* d_out, int out_size, void* d_ws, size_t ws_size,
                              hipStream_t stream) {
  const float* x   = (const float*)d_in[0];   // [4,1024,1024]
  const float* y   = (const float*)d_in[1];   // [4,1024,1024]
  const float* wts = (const float*)d_in[2];   // [4,1,1024,1024]
  const float* Wq  = (const float*)d_in[3];   // [1024,1024]
  const float* Wkv = (const float*)d_in[4];   // [2048,1024]
  const float* Wp  = (const float*)d_in[5];   // [1024,1024]
  const float* bp  = (const float*)d_in[6];   // [1024]

  float* out_f32  = (float*)d_out;                                  // 4M elems
  float* attn_f32 = out_f32 + (size_t)BATCH * NX * DIM;             // 64M elems

  // Workspace carve-up (bf16 as u16), total 28M elems = 56 MB.
  u16* ws     = (u16*)d_ws;
  u16* x_bf   = ws;                       // 4M
  u16* y_bf   = x_bf   + (4u << 20);      // 4M
  u16* wq_bf  = y_bf   + (4u << 20);      // 1M
  u16* wkv_bf = wq_bf  + (1u << 20);      // 2M
  u16* wp_bf  = wkv_bf + (2u << 20);      // 1M
  u16* q_bf   = wp_bf  + (1u << 20);      // 4M
  u16* k_bf   = q_bf   + (4u << 20);      // 4M
  u16* vT_bf  = k_bf   + (4u << 20);      // 4M
  u16* o_bf   = vT_bf  + (4u << 20);      // 4M

  const int T = 256;
  cvt_f32_bf16<<<(4 << 20) / T, T, 0, stream>>>(x,   x_bf,   4 << 20);
  cvt_f32_bf16<<<(4 << 20) / T, T, 0, stream>>>(y,   y_bf,   4 << 20);
  cvt_f32_bf16<<<(1 << 20) / T, T, 0, stream>>>(Wq,  wq_bf,  1 << 20);
  cvt_f32_bf16<<<(2 << 20) / T, T, 0, stream>>>(Wkv, wkv_bf, 2 << 20);
  cvt_f32_bf16<<<(1 << 20) / T, T, 0, stream>>>(Wp,  wp_bf,  1 << 20);

  qproj_kernel  <<<256, 256, 0, stream>>>(x_bf, wq_bf, q_bf);
  kvproj_kernel <<<512, 256, 0, stream>>>(y_bf, wkv_bf, k_bf, vT_bf);
  attn_kernel   <<<4096, 256, 0, stream>>>(q_bf, k_bf, vT_bf, wts,
                                           attn_f32, o_bf);
  outproj_kernel<<<256, 256, 0, stream>>>(o_bf, wp_bf, bp, out_f32);
}
